// MathReasoningModule_12979391168561
// MI455X (gfx1250) — matrix operand order repaired and checked by reading, not compiled, not run on hardware
//
#include <hip/hip_runtime.h>
#include <math.h>

// ---------------------------------------------------------------------------
// MathReasoningModule on MI455X (gfx1250, wave32, WMMA)
//   H=1024, B=16384, 8 steps.
//   bf16 WMMA GEMMs, GRU r/z folded, fused epilogue + next-step A-packing.
//   Two-level tiling: WG = 128 rows x 16 cols x 6 weight groups, K in 8
//   chunks of 128. A+B double-buffered in LDS (112 KB), filled with
//   GLOBAL_LOAD_ASYNC_TO_LDS_B128 (ASYNCcnt) -- no VGPR staging, no spills.
// ---------------------------------------------------------------------------

typedef __attribute__((ext_vector_type(16))) __bf16 v16bf;
typedef __attribute__((ext_vector_type(8)))  float  v8f;

#define H      1024
#define B      16384
#define NSTEPS 8
#define NTOT   6144   // [r | z | i_n | h_n | op1 | num1] each H wide

// round-to-nearest-even f32 -> bf16
__device__ __forceinline__ unsigned short f2bf(float f) {
    union { float f; unsigned int u; } c; c.f = f;
    unsigned int u = c.u;
    unsigned int r = u + 0x7FFFu + ((u >> 16) & 1u);
    return (unsigned short)(r >> 16);
}

__device__ __forceinline__ float sigmoidf_(float x) { return 1.0f / (1.0f + __expf(-x)); }
__device__ __forceinline__ float gelu_exact(float x) {
    return 0.5f * x * (1.0f + erff(x * 0.70710678118654752f));
}

// --- WMMA fragment maps (wave32, 16-bit A 16x32 / B 32x16) ------------------
// A fwd: lane L (m=L%16, hf=L/16), elem e: r=e/2,hb=e&1;
//        k = (r<4) ? hf*8+2r+hb : 16+hf*8+2(r-4)+hb
// A inv: given k (0..31): hf=(k>>3)&1 ; e=(k&7)+((k>>4)<<3)
// B:     lane L holds col n=L%16; elem e: k=(L/16)*8+(e&7)+(e>>3)*16 (same as A)
// Packed tile storage (A and B identical tiling): tile, kblock of 32:
//        flat = tile*16384 + kb*512 + lane*16 + e     (ushort units)

// ---------------------------------------------------------------------------
__global__ void prep_weights(const float* __restrict__ w_ih,
                             const float* __restrict__ w_hh,
                             const float* __restrict__ w_op1,
                             const float* __restrict__ w_num1,
                             unsigned short* __restrict__ Wt) {
    const size_t total = (size_t)NTOT * H;
    size_t stride = (size_t)gridDim.x * blockDim.x;
    for (size_t idx = (size_t)blockIdx.x * blockDim.x + threadIdx.x; idx < total; idx += stride) {
        int nt   = (int)(idx >> 14);
        int rem  = (int)(idx & 16383);
        int kb   = rem >> 9;
        int lane = (rem >> 4) & 31;
        int e    = rem & 15;
        int n = nt * 16 + (lane & 15);
        int k = kb * 32 + (lane >> 4) * 8 + (e & 7) + ((e >> 3) << 4);
        float w;
        if (n < 2 * H) {                  // r, z gates: (w_ih + w_hh) row n
            w = w_ih[(size_t)n * H + k] + w_hh[(size_t)n * H + k];
        } else if (n < 3 * H) {           // i_n
            w = w_ih[(size_t)n * H + k];
        } else if (n < 4 * H) {           // h_n
            w = w_hh[(size_t)(n - H) * H + k];
        } else if (n < 5 * H) {           // op1
            w = w_op1[(size_t)k * H + (n - 4 * H)];
        } else {                          // num1
            w = w_num1[(size_t)k * H + (n - 5 * H)];
        }
        Wt[idx] = f2bf(w);
    }
}

__global__ void prep_bias(const float* __restrict__ b_ih,
                          const float* __restrict__ b_hh,
                          const float* __restrict__ b_op1,
                          const float* __restrict__ b_num1,
                          float* __restrict__ biasv) {
    int n = blockIdx.x * blockDim.x + threadIdx.x;
    if (n >= NTOT) return;
    float b;
    if (n < 2 * H)       b = b_ih[n] + b_hh[n];
    else if (n < 3 * H)  b = b_ih[n];
    else if (n < 4 * H)  b = b_hh[n - H];
    else if (n < 5 * H)  b = b_op1[n - 4 * H];
    else                 b = b_num1[n - 5 * H];
    biasv[n] = b;
}

// read x ONCE: write states[0] (f32) and packed bf16 A tiles for step 0
__global__ void init_h0(const float* __restrict__ x,
                        float* __restrict__ s0,
                        unsigned short* __restrict__ hA) {
    const size_t total = (size_t)B * H;
    size_t stride = (size_t)gridDim.x * blockDim.x;
    for (size_t i = (size_t)blockIdx.x * blockDim.x + threadIdx.x; i < total; i += stride) {
        float v = x[i];
        s0[i] = v;
        int row = (int)(i >> 10);
        int k   = (int)(i & 1023);
        int rt = row >> 4, m = row & 15;
        int kb = k >> 5,  kl = k & 31;
        int hf = (kl >> 3) & 1;
        int e  = (kl & 7) + ((kl >> 4) << 3);
        hA[(size_t)rt * 16384 + kb * 512 + (m + 16 * hf) * 16 + e] = f2bf(v);
    }
}

// seed ops/nums with head biases (atomics accumulate partial dot-products)
__global__ void init_heads(const float* __restrict__ b_op2, const float* __restrict__ b_num2,
                           float* __restrict__ ops, float* __restrict__ nums) {
    int i = blockIdx.x * blockDim.x + threadIdx.x;
    if (i >= NSTEPS * B) return;
    #pragma unroll
    for (int c = 0; c < 5; ++c) ops[(size_t)i * 5 + c] = b_op2[c];
    nums[i] = b_num2[0];
}

// ---------------------------------------------------------------------------
// Fused GRU step, LDS-staged GEMM with async global->LDS copies.
//   grid (B/128, H/16), 256 threads = 8 wave32; wave w owns row-tile w.
//   K processed in 8 chunks of 4 K-blocks (128 K values).
//   LDS chunk layout (1KB blocks): blocks 0..31 = A (rt*4+kb),
//   blocks 32..55 = B (32 + g*4 + kb). 56 KB per buffer, ping-pong.
//   Each wave issues 14 GLOBAL_LOAD_ASYNC_TO_LDS_B128 per chunk
//   (7 blocks x 2 half-KB, 512 B per instruction across 32 lanes).
// ---------------------------------------------------------------------------
#define KCH    4                 // K-blocks per chunk
#define NCH    (32 / KCH)        // 8 chunks
#define CHBLK  (8 * KCH + 6 * KCH)   // 56 blocks of 1KB
#define CHBYTE (CHBLK * 1024)        // 57344 B per buffer

__global__ __launch_bounds__(256) void gru_fused_step(
    const unsigned short* __restrict__ Wt,
    const unsigned short* __restrict__ hA,
    const float* __restrict__ biasv,
    const float* __restrict__ h_in,
    const float* __restrict__ w_op2,   // [H,5]
    const float* __restrict__ w_num2,  // [H]
    float* __restrict__ h_out,
    unsigned short* __restrict__ hA_next,  // packed bf16 h_new (or null)
    float* __restrict__ opsT,          // [B,5]
    float* __restrict__ numsT)         // [B]
{
    __shared__ __align__(16) unsigned short lds[2][CHBLK * 512];  // 2 x 56 KB

    const int tid  = threadIdx.x;
    const int wave = tid >> 5;
    const int lane = tid & 31;
    const int half = lane >> 4;
    const int j0   = blockIdx.y * 16;
    const int jc   = j0 + (lane & 15);

    // raw LDS byte offset of the buffers (low 32 bits of generic LDS address)
    const unsigned ldsbase = (unsigned)(unsigned long long)(&lds[0][0]);

    // issue one chunk's async copies: wave w covers blocks [w*7, w*7+7)
    auto issue_chunk = [&](int c, int bufsel) {
        #pragma unroll
        for (int i = 0; i < 14; ++i) {
            int a   = wave * 14 + i;          // 0..111
            int blk = a >> 1;                 // 1KB block 0..55
            int hf2 = a & 1;                  // half-KB within block
            unsigned long long src;
            if (blk < 32) {
                int rt = blk >> 2, kbl = blk & 3;
                src = (unsigned long long)(const char*)hA
                    + (unsigned long long)(blockIdx.x * 8 + rt) * 32768ull
                    + (unsigned long long)(c * KCH + kbl) * 1024ull;
            } else {
                int g = (blk - 32) >> 2, kbl = blk & 3;
                src = (unsigned long long)(const char*)Wt
                    + (unsigned long long)(g * H + j0) * 2048ull
                    + (unsigned long long)(c * KCH + kbl) * 1024ull;
            }
            src += (unsigned long long)(hf2 * 512 + lane * 16);
            unsigned dst = ldsbase + (unsigned)(bufsel * CHBYTE + blk * 1024 + hf2 * 512 + lane * 16);
            asm volatile("global_load_async_to_lds_b128 %0, %1, off"
                         :: "v"(dst), "v"(src) : "memory");
        }
    };

    // ---- prologue: stage chunk 0
    issue_chunk(0, 0);
    asm volatile("s_wait_asynccnt 0x0" ::: "memory");
    __syncthreads();

    v8f aR  = {0.f,0.f,0.f,0.f,0.f,0.f,0.f,0.f};
    v8f aZ  = aR, aIN = aR, aHN = aR, aOP = aR, aNU = aR;

    for (int c = 0; c < NCH; ++c) {
        if (c + 1 < NCH) issue_chunk(c + 1, (c + 1) & 1);   // overlap with compute
        // ---- compute on chunk c from LDS
        const unsigned short* pAw = &lds[c & 1][0] + lane * 16;   // + block*512
        #pragma unroll
        for (int kb = 0; kb < KCH; ++kb) {
            v16bf a  = *reinterpret_cast<const v16bf*>(pAw + (wave * KCH + kb) * 512);
            v16bf b0 = *reinterpret_cast<const v16bf*>(pAw + (32 + 0 * KCH + kb) * 512);
            v16bf b1 = *reinterpret_cast<const v16bf*>(pAw + (32 + 1 * KCH + kb) * 512);
            v16bf b2 = *reinterpret_cast<const v16bf*>(pAw + (32 + 2 * KCH + kb) * 512);
            aR  = __builtin_amdgcn_wmma_f32_16x16x32_bf16(false, a, false, b0, (short)0, aR,  false, false);
            aZ  = __builtin_amdgcn_wmma_f32_16x16x32_bf16(false, a, false, b1, (short)0, aZ,  false, false);
            aIN = __builtin_amdgcn_wmma_f32_16x16x32_bf16(false, a, false, b2, (short)0, aIN, false, false);
            v16bf b3 = *reinterpret_cast<const v16bf*>(pAw + (32 + 3 * KCH + kb) * 512);
            v16bf b4 = *reinterpret_cast<const v16bf*>(pAw + (32 + 4 * KCH + kb) * 512);
            v16bf b5 = *reinterpret_cast<const v16bf*>(pAw + (32 + 5 * KCH + kb) * 512);
            aHN = __builtin_amdgcn_wmma_f32_16x16x32_bf16(false, a, false, b3, (short)0, aHN, false, false);
            aOP = __builtin_amdgcn_wmma_f32_16x16x32_bf16(false, a, false, b4, (short)0, aOP, false, false);
            aNU = __builtin_amdgcn_wmma_f32_16x16x32_bf16(false, a, false, b5, (short)0, aNU, false, false);
        }
        if (c + 1 < NCH) {
            asm volatile("s_wait_asynccnt 0x0" ::: "memory");   // chunk c+1 landed
            __syncthreads();                                    // publish to all waves
        }
    }

    // epilogue: GRU blend + heads. C/D layout: lane 0-15 -> N=lane, M=v;
    // lane 16-31 -> N=lane-16, M=v+8. Wave w owns rows [bx*128+w*16, +16).
    const float bR_  = biasv[0 * H + jc];
    const float bZ_  = biasv[1 * H + jc];
    const float bIN_ = biasv[2 * H + jc];
    const float bHN_ = biasv[3 * H + jc];
    const float bOP_ = biasv[4 * H + jc];
    const float bNU_ = biasv[5 * H + jc];
    float w2[5];
    #pragma unroll
    for (int c = 0; c < 5; ++c) w2[c] = w_op2[(size_t)jc * 5 + c];
    const float wn = w_num2[jc];

    // inverse A-fragment map for this lane's column jc
    const int kb_p = jc >> 5;
    const int kl_p = jc & 31;
    const int hf_p = (kl_p >> 3) & 1;
    const int e_p  = (kl_p & 7) + ((kl_p >> 4) << 3);

    #pragma unroll
    for (int v = 0; v < 8; ++v) {
        const int grow = blockIdx.x * 128 + wave * 16 + v + half * 8;
        const size_t hidx = (size_t)grow * H + jc;

        float r = sigmoidf_(aR[v] + bR_);
        float z = sigmoidf_(aZ[v] + bZ_);
        float n = tanhf((aIN[v] + bIN_) + r * (aHN[v] + bHN_));
        float hold = h_in[hidx];
        float hnew = (1.0f - z) * n + z * hold;
        h_out[hidx] = hnew;

        if (hA_next) {
            const int rt = grow >> 4, m = grow & 15;
            hA_next[(size_t)rt * 16384 + kb_p * 512 + (m + 16 * hf_p) * 16 + e_p] = f2bf(hnew);
        }

        float uop = gelu_exact(aOP[v] + bOP_);
        float unm = gelu_exact(aNU[v] + bNU_);

        float p[6];
        #pragma unroll
        for (int c = 0; c < 5; ++c) p[c] = uop * w2[c];
        p[5] = unm * wn;
        #pragma unroll
        for (int m2 = 1; m2 < 16; m2 <<= 1) {
            #pragma unroll
            for (int c = 0; c < 6; ++c) p[c] += __shfl_xor(p[c], m2, 32);
        }
        if ((lane & 15) == 0) {
            #pragma unroll
            for (int c = 0; c < 5; ++c) atomicAdd(&opsT[(size_t)grow * 5 + c], p[c]);
            atomicAdd(&numsT[grow], p[5]);
        }
    }
}

// ---------------------------------------------------------------------------
extern "C" void kernel_launch(void* const* d_in, const int* in_sizes, int n_in,
                              void* d_out, int out_size, void* d_ws, size_t ws_size,
                              hipStream_t stream) {
    const float* x      = (const float*)d_in[0];
    const float* w_op1  = (const float*)d_in[2];
    const float* b_op1  = (const float*)d_in[3];
    const float* w_op2  = (const float*)d_in[4];
    const float* b_op2  = (const float*)d_in[5];
    const float* w_num1 = (const float*)d_in[6];
    const float* b_num1 = (const float*)d_in[7];
    const float* w_num2 = (const float*)d_in[8];
    const float* b_num2 = (const float*)d_in[9];
    const float* w_ih   = (const float*)d_in[10];
    const float* b_ih   = (const float*)d_in[11];
    const float* w_hh   = (const float*)d_in[12];
    const float* b_hh   = (const float*)d_in[13];
    float* out = (float*)d_out;

    // output layout (flat, return order)
    const size_t OPS_OFF   = (size_t)B * H;                     // ops [8,B,5]
    const size_t NUMS_OFF  = OPS_OFF + (size_t)NSTEPS * B * 5;  // nums [8,B,1]
    const size_t ST_OFF    = NUMS_OFF + (size_t)NSTEPS * B;     // states [8,B,H]
    float* fin    = out;
    float* ops    = out + OPS_OFF;
    float* nums   = out + NUMS_OFF;
    float* states = out + ST_OFF;

    // workspace layout: Wt (12 MB) | hA ping (32 MB) | hA pong (32 MB) | bias
    unsigned short* Wt  = (unsigned short*)d_ws;                // NTOT*H bf16
    unsigned short* hA0 = Wt + (size_t)NTOT * H;                // B*H bf16 packed A
    unsigned short* hA1 = hA0 + (size_t)B * H;
    float*          biasv = (float*)(hA1 + (size_t)B * H);      // NTOT f32

    prep_weights<<<4096, 256, 0, stream>>>(w_ih, w_hh, w_op1, w_num1, Wt);
    prep_bias   <<<(NTOT + 255) / 256, 256, 0, stream>>>(b_ih, b_hh, b_op1, b_num1, biasv);
    init_h0     <<<8192, 256, 0, stream>>>(x, states, hA0);     // states[0]=x + packed A
    init_heads  <<<(NSTEPS * B + 255) / 256, 256, 0, stream>>>(b_op2, b_num2, ops, nums);

    for (int t = 0; t < NSTEPS; ++t) {
        const float* hin = states + (size_t)t * B * H;
        float* hout = (t + 1 < NSTEPS) ? (states + (size_t)(t + 1) * B * H) : fin;
        unsigned short* hAcur = (t & 1) ? hA1 : hA0;
        unsigned short* hAnxt = (t + 1 < NSTEPS) ? ((t & 1) ? hA0 : hA1) : (unsigned short*)nullptr;
        dim3 grid(B / 128, H / 16);
        gru_fused_step<<<grid, 256, 0, stream>>>(Wt, hAcur, biasv, hin, w_op2, w_num2,
                                                 hout, hAnxt,
                                                 ops + (size_t)t * B * 5,
                                                 nums + (size_t)t * B);
    }
}
